// SGAM_5102421147751
// MI455X (gfx1250) — matrix operand order repaired and checked by reading, not compiled, not run on hardware
//
#include <hip/hip_runtime.h>
#include <hip/hip_bf16.h>

// Problem constants (B,C,H,W) = (2,256,64,64)
#define N_NODES 8192          // B*H*W
#define C_DIM   256
#define NELEM   (N_NODES * C_DIM)
#define ROWB    (C_DIM * 2)   // bytes per Y row (bf16) = 512

#define IB_PER_WG 8                         // 8 waves -> 8 i-blocks per workgroup
#define JC        32                        // j-chunks (column parallelism)
#define JBLK_TOTAL (N_NODES / 16)           // 512 column blocks
#define JB_PER_CHUNK (JBLK_TOTAL / JC)      // 16 column blocks per chunk

typedef __attribute__((ext_vector_type(16))) __bf16 v16bf;
typedef __attribute__((ext_vector_type(8)))  float  v8f;

__device__ __forceinline__ int iabs(int v) { return v < 0 ? -v : v; }

// --- CDNA5 async global->LDS copy (ASYNCcnt path), inline asm ------------
__device__ __forceinline__ void async_copy16(unsigned lds_off, const void* g) {
    asm volatile("global_load_async_to_lds_b128 %0, %1, off"
                 :: "v"(lds_off), "v"(g) : "memory");
}
__device__ __forceinline__ void wait_async0() {
#if defined(__has_builtin) && __has_builtin(__builtin_amdgcn_s_wait_asynccnt)
    __builtin_amdgcn_s_wait_asynccnt(0);
#else
    asm volatile("s_wait_asynccnt 0x0" ::: "memory");
#endif
}

// ---------------------------------------------------------------------------
// Kernel 1: row norms + normalized bf16 copy.
// X[n,c] = raw[c*N + n];  Y[n*C + c] = bf16(X[n,c] / max(||X[n]||, 1e-8))
// ---------------------------------------------------------------------------
__global__ void __launch_bounds__(256) norm_kernel(const float* __restrict__ x,
                                                   __hip_bfloat16* __restrict__ Y) {
    const int n = blockIdx.x * blockDim.x + threadIdx.x;   // coalesced over n
    float s = 0.f;
    for (int c = 0; c < C_DIM; ++c) {
        float v = x[(size_t)c * N_NODES + n];
        s += v * v;
    }
    float inv = 1.0f / fmaxf(sqrtf(s), 1e-8f);
    for (int c = 0; c < C_DIM; ++c) {
        float v = x[(size_t)c * N_NODES + n] * inv;
        Y[(size_t)n * C_DIM + c] = __float2bfloat16(v);
    }
}

// ---------------------------------------------------------------------------
// Kernel 2: fused Gram + exp-row-sum.
// Workgroup = 8 waves; wave w owns i-block (gi*8+w), A panel in registers.
// All waves share one j-chunk; each 16x256 B panel (8KB) is async-staged
// into double-buffered LDS once per WG and read back as ds_load_b128 pairs.
// ---------------------------------------------------------------------------
__global__ void __launch_bounds__(256) gram_kernel(const unsigned int* __restrict__ Yw,
                                                   float* __restrict__ rowsum,
                                                   float* __restrict__ diag) {
    __shared__ __align__(16) unsigned char smem[2][16 * ROWB];   // 2 x 8KB

    const int t    = threadIdx.x;        // 0..255
    const int lane = t & 31;             // wave32 lane
    const int wave = t >> 5;             // 0..7
    const int half = lane >> 4;          // 0/1 lane half
    const int mrow = lane & 15;

    const int gi    = blockIdx.x / JC;   // i-block group (0..63)
    const int chunk = blockIdx.x % JC;
    const int i_blk = gi * IB_PER_WG + wave;
    const int i_base = i_blk * 16;

    const unsigned char* Yb = (const unsigned char*)Yw;
    const unsigned lds_base = (unsigned)(unsigned long long)(const void*)&smem[0][0];

    // --- Preload A panel: this wave's 16 rows, all 8 K-slices (registers) --
    // A layout (16-bit 16x32): lane M = lane&15,
    //   VGPR q holds K pair starting at 2q + 16*half (contiguous, matches B).
    v16bf a[8];
    {
        const int rowA = i_base + mrow;
        #pragma unroll
        for (int s = 0; s < 8; ++s) {
            union { v16bf v; unsigned int u[8]; } ta;
            #pragma unroll
            for (int q = 0; q < 8; ++q) {
                int kb = 2 * q + 16 * half;
                int k  = s * 32 + kb;                   // even -> dword aligned
                ta.u[q] = Yw[rowA * (C_DIM / 2) + (k >> 1)];
            }
            a[s] = ta.v;
        }
    }

    // Output-row (b,h,w) coords (C/D layout: M = r + 8*half)
    int bi[8], hi[8], wi[8];
    #pragma unroll
    for (int r = 0; r < 8; ++r) {
        int I = i_base + r + 8 * half;
        bi[r] = I >> 12; hi[r] = (I >> 6) & 63; wi[r] = I & 63;
    }

    float facc[8];
    #pragma unroll
    for (int r = 0; r < 8; ++r) facc[r] = 0.f;

    // Staging role of this thread: 16 rows x 16 segments of 32B = 8KB panel.
    const int srow = t >> 4;             // 0..15
    const int sseg = t & 15;             // 0..15

    const int jb0  = chunk * JB_PER_CHUNK;
    const int jend = jb0 + JB_PER_CHUNK;

    // Prologue: stage first panel.
    {
        const unsigned char* g = Yb + (size_t)(jb0 * 16 + srow) * ROWB + sseg * 32;
        unsigned l = lds_base + srow * ROWB + sseg * 32;
        async_copy16(l, g);
        async_copy16(l + 16, g + 16);
    }
    wait_async0();
    __syncthreads();

    int p = 0;
    for (int jb = jb0; jb < jend; ++jb) {
        // Stage next panel into the other buffer while computing this one.
        if (jb + 1 < jend) {
            const unsigned char* g = Yb + (size_t)((jb + 1) * 16 + srow) * ROWB + sseg * 32;
            unsigned l = lds_base + (p ^ 1) * (16 * ROWB) + srow * ROWB + sseg * 32;
            async_copy16(l, g);
            async_copy16(l + 16, g + 16);
        }

        const int j_base = jb * 16;
        const int J  = j_base + mrow;                   // this lane's column N
        const int bj = J >> 12, hj = (J >> 6) & 63, wj = J & 63;

        // B layout (16-bit 32x16): lane N = lane&15; lane's 8 dwords for slice
        // s are contiguous: byte offset = N*512 + 64*s + 32*half  (32 bytes).
        const unsigned char* bpan = &smem[p][0] + mrow * ROWB + 32 * half;

        v8f c = {0.f, 0.f, 0.f, 0.f, 0.f, 0.f, 0.f, 0.f};
        #pragma unroll
        for (int s = 0; s < 8; ++s) {
            union { v16bf v; uint4 u4[2]; } tb;
            tb.u4[0] = *(const uint4*)(bpan + 64 * s);
            tb.u4[1] = *(const uint4*)(bpan + 64 * s + 16);
            c = __builtin_amdgcn_wmma_f32_16x16x32_bf16(
                    /*neg_a=*/false, a[s], /*neg_b=*/false, tb.v,
                    /*c_mod=*/(short)0, c, /*reuse_a=*/false, /*reuse_b=*/false);
        }

        // Epilogue: adjacency + exp accumulation (co-executes with XDL).
        #pragma unroll
        for (int r = 0; r < 8; ++r) {
            const int I = i_base + r + 8 * half;
            const int dist = iabs(bi[r] - bj) + iabs(hi[r] - hj) + iabs(wi[r] - wj);
            const bool dg = (I == J);
            const float adj = (dist == 1 || dg) ? 0.5f : 0.0f;  // A in {0,1}
            const float m = 0.5f * c[r] + adj;
            facc[r] += __expf(m);
            if (dg) diag[I] = m;                        // exactly one writer per I
        }

        wait_async0();        // own staging landed in LDS
        __syncthreads();      // all waves done staging & reading
        p ^= 1;
    }

    // Reduce row partials across each 16-lane half, combine globally.
    #pragma unroll
    for (int r = 0; r < 8; ++r) {
        float v = facc[r];
        v += __shfl_xor(v, 1, 16);
        v += __shfl_xor(v, 2, 16);
        v += __shfl_xor(v, 4, 16);
        v += __shfl_xor(v, 8, 16);
        if (mrow == 0) atomicAdd(&rowsum[i_base + r + 8 * half], v);
    }
}

// ---------------------------------------------------------------------------
// Kernel 3: d[n] = exp(m_nn) / rowsum[n]
// ---------------------------------------------------------------------------
__global__ void __launch_bounds__(256) dvec_kernel(const float* __restrict__ diag,
                                                   const float* __restrict__ rowsum,
                                                   float* __restrict__ dvec) {
    const int n = blockIdx.x * blockDim.x + threadIdx.x;
    dvec[n] = __expf(diag[n]) / rowsum[n];
}

// ---------------------------------------------------------------------------
// Kernel 4: out_raw[c*N + n] = relu( x_raw[c*N + n] * d[n] )
// ---------------------------------------------------------------------------
__global__ void __launch_bounds__(256) out_kernel(const float* __restrict__ x,
                                                  const float* __restrict__ dvec,
                                                  float* __restrict__ out) {
    const int idx = blockIdx.x * blockDim.x + threadIdx.x;
    const int n = idx & (N_NODES - 1);                 // coalesced: idx = c*N + n
    float v = x[idx] * dvec[n];
    out[idx] = v > 0.f ? v : 0.f;
}

// ---------------------------------------------------------------------------
extern "C" void kernel_launch(void* const* d_in, const int* in_sizes, int n_in,
                              void* d_out, int out_size, void* d_ws, size_t ws_size,
                              hipStream_t stream) {
    const float* x = (const float*)d_in[0];
    float* out = (float*)d_out;

    char* ws = (char*)d_ws;
    __hip_bfloat16* Y = (__hip_bfloat16*)ws;                       // 4 MB
    float* rowsum = (float*)(ws + (size_t)N_NODES * C_DIM * 2);    // 32 KB
    float* diag   = rowsum + N_NODES;                              // 32 KB
    float* dvec   = diag + N_NODES;                                // 32 KB

    hipMemsetAsync(rowsum, 0, N_NODES * sizeof(float), stream);

    norm_kernel<<<N_NODES / 256, 256, 0, stream>>>(x, Y);
    gram_kernel<<<(N_NODES / 16 / IB_PER_WG) * JC, 256, 0, stream>>>(
        (const unsigned int*)Y, rowsum, diag);
    dvec_kernel<<<N_NODES / 256, 256, 0, stream>>>(diag, rowsum, dvec);
    out_kernel<<<NELEM / 256, 256, 0, stream>>>(x, dvec, out);
}
